// HedgehogAttention_25675314495619
// MI455X (gfx1250) — matrix operand order repaired and checked by reading, not compiled, not run on hardware
//
#include <hip/hip_runtime.h>

// ---------------------------------------------------------------------------
// MI455X (gfx1250) Hedgehog attention, wave32 + WMMA bf16 (f32 accumulate).
// All heavy matmuls on v_wmma_f32_16x16x32_bf16; flash-style attention so the
// 16x3072x3072 logits never hit memory. Per-query spiky scale folded into Q.
// Staging uses packed bf16 (v_cvt_pk_bf16_f32) + 128-bit LDS/global accesses.
// ---------------------------------------------------------------------------

typedef __attribute__((ext_vector_type(16))) __bf16 bf16x16;
typedef __attribute__((ext_vector_type(8)))  float  f32x8;

#define N_TOK    3072
#define DMODEL   1024
#define NHEADS   16
#define DHEAD    64
#define QKV_COLS 3072
#define ATT_SCALE 0.125f   // 64^-0.5

struct alignas(16) U4 { unsigned x, y, z, w; };
union BF2 { unsigned u; __bf16 b[2]; };

__device__ __forceinline__ unsigned pkbf2(float a, float b) {
  BF2 p; p.b[0] = (__bf16)a; p.b[1] = (__bf16)b; return p.u;   // v_cvt_pk_bf16_f32
}
__device__ __forceinline__ U4 cvt8(const float* __restrict__ s) {
  U4 r; r.x = pkbf2(s[0], s[1]); r.y = pkbf2(s[2], s[3]);
  r.z = pkbf2(s[4], s[5]); r.w = pkbf2(s[6], s[7]); return r;
}

__device__ __forceinline__ f32x8 wmma_bf16f32(bf16x16 a, bf16x16 b, f32x8 c) {
  // (neg_a, A, neg_b, B, c_mod, C, reuse_a, reuse_b)
  return __builtin_amdgcn_wmma_f32_16x16x32_bf16(false, a, false, b, (short)0, c,
                                                 false, false);
}

// ---------------------------------------------------------------------------
// Kernel 1: qkv = x @ Wqkv  -> q/k/v stored bf16, head-major [h][n][d]
// Grid (48, 48): 64x64 output tile per WG; 128 thr = 4 waves, each wave owns a
// 16-wide N strip over 64 rows (4 WMMA accumulators).
// ---------------------------------------------------------------------------
__global__ void __launch_bounds__(128)
qkv_gemm_kernel(const float* __restrict__ X, const float* __restrict__ W,
                __bf16* __restrict__ qkv) {
  __shared__ alignas(16) __bf16 Xs[64][32];   // [m][k]
  __shared__ alignas(16) __bf16 Ws[32][64];   // [k][n]
  const int t = threadIdx.x;
  const int wv = t >> 5, lane = t & 31, half = lane >> 4, l16 = lane & 15;
  const int m0 = blockIdx.x * 64;
  const int n0 = blockIdx.y * 64;

  f32x8 acc[4] = {};

  for (int k0 = 0; k0 < DMODEL; k0 += 32) {
    for (int i = 0; i < 2; i++) {              // X tile 64x32: 8-elem chunks
      int chunk = t + i * 128;                 // 0..255
      int r = chunk >> 2, c8 = (chunk & 3) * 8;
      *(U4*)&Xs[r][c8] = cvt8(&X[(size_t)(m0 + r) * DMODEL + k0 + c8]);
    }
    for (int i = 0; i < 2; i++) {              // W tile 32x64: 8-elem chunks
      int chunk = t + i * 128;
      int r = chunk >> 3, c8 = (chunk & 7) * 8;
      *(U4*)&Ws[r][c8] = cvt8(&W[(size_t)(k0 + r) * QKV_COLS + n0 + c8]);
    }
    __syncthreads();

    bf16x16 bfrag;                              // B: lane = N col, e -> K map
    for (int e = 0; e < 8; e++) bfrag[e]     = Ws[8 * half + e][wv * 16 + l16];
    for (int e = 0; e < 8; e++) bfrag[e + 8] = Ws[16 + 8 * half + e][wv * 16 + l16];

    for (int mt = 0; mt < 4; mt++) {
      bf16x16 afrag;                            // A: lane = M, e -> K map
      int row = mt * 16 + l16;
      for (int e = 0; e < 8; e++) afrag[e]     = Xs[row][8 * half + e];
      for (int e = 0; e < 8; e++) afrag[e + 8] = Xs[row][16 + 8 * half + e];
      acc[mt] = wmma_bf16f32(afrag, bfrag, acc[mt]);
    }
    __syncthreads();
  }

  // Route tile (64-aligned cols => single section + single head per WG).
  const int section = n0 / DMODEL;              // 0=q 1=k 2=v
  const int head    = (n0 % DMODEL) / DHEAD;
  __bf16* base = qkv + (size_t)section * NHEADS * N_TOK * DHEAD
                     + (size_t)head * N_TOK * DHEAD;
  const int dd = wv * 16 + l16;
  for (int mt = 0; mt < 4; mt++)
    for (int e = 0; e < 8; e++) {
      int row = m0 + mt * 16 + e + 8 * half;    // D: elem r -> row r+8*half
      base[(size_t)row * DHEAD + dd] = (__bf16)acc[mt][e];
    }
}

// ---------------------------------------------------------------------------
// Kernel 2: spiky = sigmoid(relu(k@W1+b1)@W2+b2); softmax over n per head.
// Output wscale[h][n] = ATT_SCALE * softmax weight (folded into Q later).
// ---------------------------------------------------------------------------
__global__ void __launch_bounds__(256)
spiky_kernel(const __bf16* __restrict__ kbuf,
             const float* __restrict__ W1, const float* __restrict__ b1,
             const float* __restrict__ W2, const float* __restrict__ b2,
             float* __restrict__ wscale) {
  __shared__ float W1s[64][64];
  __shared__ float W2s[64], b1s[64];
  __shared__ float sv[N_TOK];
  __shared__ float red[256];
  const int h = blockIdx.x, t = threadIdx.x;

  for (int i = t; i < 64 * 64; i += 256) W1s[i >> 6][i & 63] = W1[i];
  if (t < 64) { W2s[t] = W2[t]; b1s[t] = b1[t]; }
  __syncthreads();

  const __bf16* kh = kbuf + (size_t)h * N_TOK * DHEAD;
  const float b2v = b2[0];
  for (int j = t; j < N_TOK; j += 256) {
    float kv[64];
    for (int c8 = 0; c8 < 64; c8 += 8) {        // 16B bf16 chunk loads
      U4 raw = *(const U4*)&kh[(size_t)j * DHEAD + c8];
      BF2 p;
      p.u = raw.x; kv[c8 + 0] = (float)p.b[0]; kv[c8 + 1] = (float)p.b[1];
      p.u = raw.y; kv[c8 + 2] = (float)p.b[0]; kv[c8 + 3] = (float)p.b[1];
      p.u = raw.z; kv[c8 + 4] = (float)p.b[0]; kv[c8 + 5] = (float)p.b[1];
      p.u = raw.w; kv[c8 + 6] = (float)p.b[0]; kv[c8 + 7] = (float)p.b[1];
    }
    float s2 = b2v;
    for (int c = 0; c < 64; c++) {
      float a = b1s[c];
      for (int d = 0; d < 64; d++) a = fmaf(kv[d], W1s[d][c], a);
      s2 = fmaf(fmaxf(a, 0.f), W2s[c], s2);
    }
    sv[j] = 1.f / (1.f + __expf(-s2));
  }
  __syncthreads();

  float mx = -1e30f;
  for (int j = t; j < N_TOK; j += 256) mx = fmaxf(mx, sv[j]);
  red[t] = mx; __syncthreads();
  for (int s = 128; s > 0; s >>= 1) {
    if (t < s) red[t] = fmaxf(red[t], red[t + s]);
    __syncthreads();
  }
  mx = red[0]; __syncthreads();

  float sum = 0.f;
  for (int j = t; j < N_TOK; j += 256) {
    float e = __expf(sv[j] - mx); sv[j] = e; sum += e;
  }
  red[t] = sum; __syncthreads();
  for (int s = 128; s > 0; s >>= 1) {
    if (t < s) red[t] += red[t + s];
    __syncthreads();
  }
  const float inv = 1.f / red[0];
  for (int j = t; j < N_TOK; j += 256)
    wscale[h * N_TOK + j] = ATT_SCALE * sv[j] * inv;
}

// ---------------------------------------------------------------------------
// Kernel 3: flash attention per head. Grid (48, 16); 128 thr = 4 waves, each
// wave owns 16 queries. K/V streamed in 32-key tiles (L2-resident workspace).
// Online softmax with cross-lane (16-lane half) reductions matching the
// WMMA C/D layout; P bounced through wave-private LDS into A-fragment layout.
// ---------------------------------------------------------------------------
__global__ void __launch_bounds__(128)
flash_attn_kernel(const __bf16* __restrict__ qbuf, const __bf16* __restrict__ kbuf,
                  const __bf16* __restrict__ vbuf, const float* __restrict__ wscale,
                  __bf16* __restrict__ attn) {
  __shared__ alignas(16) __bf16 Qs[64][64];    // [qrow][d], pre-scaled
  __shared__ alignas(16) __bf16 Kts[64][32];   // [d][key] (transposed, B frags)
  __shared__ alignas(16) __bf16 Vs[32][64];    // [key][d]
  __shared__ alignas(16) __bf16 Ps[4][16][32]; // per-wave P bounce buffer
  const int t = threadIdx.x;
  const int wv = t >> 5, lane = t & 31, half = lane >> 4, l16 = lane & 15;
  const int h  = blockIdx.y;
  const int q0 = blockIdx.x * 64;
  const __bf16* qh = qbuf + (size_t)h * N_TOK * DHEAD;
  const __bf16* kh = kbuf + (size_t)h * N_TOK * DHEAD;
  const __bf16* vh = vbuf + (size_t)h * N_TOK * DHEAD;

  for (int i = 0; i < 4; i++) {                // stage scaled Q: 16B chunks
    int chunk = t + i * 128;                   // 0..511
    int r = chunk >> 3, c8 = (chunk & 7) * 8;
    float w = wscale[h * N_TOK + q0 + r];
    U4 raw = *(const U4*)&qh[(size_t)(q0 + r) * DHEAD + c8];
    U4 pk;
    BF2 p;
    p.u = raw.x; pk.x = pkbf2((float)p.b[0] * w, (float)p.b[1] * w);
    p.u = raw.y; pk.y = pkbf2((float)p.b[0] * w, (float)p.b[1] * w);
    p.u = raw.z; pk.z = pkbf2((float)p.b[0] * w, (float)p.b[1] * w);
    p.u = raw.w; pk.w = pkbf2((float)p.b[0] * w, (float)p.b[1] * w);
    *(U4*)&Qs[r][c8] = pk;
  }
  __syncthreads();

  bf16x16 aq[2];                               // Q A-frags (K=64 -> 2 chunks)
  {
    int row = wv * 16 + l16;
    for (int kc = 0; kc < 2; kc++) {
      for (int e = 0; e < 8; e++) aq[kc][e]     = Qs[row][kc * 32 + 8 * half + e];
      for (int e = 0; e < 8; e++) aq[kc][e + 8] = Qs[row][kc * 32 + 16 + 8 * half + e];
    }
  }
  __syncthreads();                             // Qs no longer needed in LDS

  f32x8 acc[4] = {};
  float m8[8], l8[8];
  for (int e = 0; e < 8; e++) { m8[e] = -1e30f; l8[e] = 0.f; }

  for (int j0 = 0; j0 < N_TOK; j0 += 32) {
    for (int i = 0; i < 2; i++) {              // stage K^T (scatter) + V (copy)
      int chunk = t + i * 128;                 // 0..255
      int key = chunk >> 3, d8 = (chunk & 7) * 8;
      U4 kraw = *(const U4*)&kh[(size_t)(j0 + key) * DHEAD + d8];
      BF2 p;
      p.u = kraw.x; Kts[d8 + 0][key] = p.b[0]; Kts[d8 + 1][key] = p.b[1];
      p.u = kraw.y; Kts[d8 + 2][key] = p.b[0]; Kts[d8 + 3][key] = p.b[1];
      p.u = kraw.z; Kts[d8 + 4][key] = p.b[0]; Kts[d8 + 5][key] = p.b[1];
      p.u = kraw.w; Kts[d8 + 6][key] = p.b[0]; Kts[d8 + 7][key] = p.b[1];
      *(U4*)&Vs[key][d8] = *(const U4*)&vh[(size_t)(j0 + key) * DHEAD + d8];
    }
    if (j0 + 32 < N_TOK) {                     // gfx1250 global_prefetch_b8
      __builtin_prefetch(&kh[(size_t)(j0 + 32) * DHEAD], 0, 1);
      __builtin_prefetch(&vh[(size_t)(j0 + 32) * DHEAD], 0, 1);
    }
    __syncthreads();

    // S = Q' K^T : two 16x16 tiles (keys j0..+15, j0+16..+31)
    f32x8 s0 = {}, s1 = {};
    for (int kc = 0; kc < 2; kc++) {
      bf16x16 bk0, bk1;                        // B: lane = key col, e -> d map
      for (int e = 0; e < 8; e++) { bk0[e] = Kts[kc * 32 + 8 * half + e][l16]; bk0[e + 8] = Kts[kc * 32 + 16 + 8 * half + e][l16]; }
      for (int e = 0; e < 8; e++) { bk1[e] = Kts[kc * 32 + 8 * half + e][16 + l16]; bk1[e + 8] = Kts[kc * 32 + 16 + 8 * half + e][16 + l16]; }
      s0 = wmma_bf16f32(aq[kc], bk0, s0);
      s1 = wmma_bf16f32(aq[kc], bk1, s1);
    }

    // Online softmax update (row r = e + 8*half lives across a 16-lane half).
    for (int e = 0; e < 8; e++) {
      float rv = fmaxf(s0[e], s1[e]);
      rv = fmaxf(rv, __shfl_xor(rv, 1, 32));
      rv = fmaxf(rv, __shfl_xor(rv, 2, 32));
      rv = fmaxf(rv, __shfl_xor(rv, 4, 32));
      rv = fmaxf(rv, __shfl_xor(rv, 8, 32));
      float mnew  = fmaxf(m8[e], rv);
      float alpha = __expf(m8[e] - mnew);
      float e0 = __expf(s0[e] - mnew);
      float e1 = __expf(s1[e] - mnew);
      float rs = e0 + e1;
      rs += __shfl_xor(rs, 1, 32);
      rs += __shfl_xor(rs, 2, 32);
      rs += __shfl_xor(rs, 4, 32);
      rs += __shfl_xor(rs, 8, 32);
      l8[e] = l8[e] * alpha + rs;
      m8[e] = mnew;
      for (int dt = 0; dt < 4; dt++) acc[dt][e] *= alpha;
      int prow = e + 8 * half;                 // D layout -> row-major bounce
      ((unsigned*)&Ps[wv][prow][0])[l16] = pkbf2(e0, e1) /* cols l16*2..+1? */;
    }
    // NOTE: packed write above interleaves (e0,e1) as columns (2*l16, 2*l16+1),
    // i.e. keys are stored in interleaved order j = (c&1)*16 + (c>>1). Undo the
    // interleave when gathering the A fragment below.
    bf16x16 ap;                                // P as A-frag (16 rows x 32 keys)
    for (int e = 0; e < 8; e++) {
      int k0i = 8 * half + e;                  // logical key 0..15  -> e0 slot
      int k1i = 16 + 8 * half + e;             // logical key 16..31 -> e1 slot
      ap[e]     = Ps[wv][l16][((k0i & 15) << 1) | (k0i >> 4)];
      ap[e + 8] = Ps[wv][l16][((k1i & 15) << 1) | (k1i >> 4)];
    }

    for (int dt = 0; dt < 4; dt++) {           // O += P @ V
      bf16x16 bv;                              // B: lane = dim col, e -> key map
      for (int e = 0; e < 8; e++) bv[e]     = Vs[8 * half + e][dt * 16 + l16];
      for (int e = 0; e < 8; e++) bv[e + 8] = Vs[16 + 8 * half + e][dt * 16 + l16];
      acc[dt] = wmma_bf16f32(ap, bv, acc[dt]);
    }
    __syncthreads();
  }

  for (int dt = 0; dt < 4; dt++)               // normalize + store
    for (int e = 0; e < 8; e++) {
      int row = q0 + wv * 16 + e + 8 * half;
      int col = dt * 16 + l16;
      attn[(size_t)row * DMODEL + h * DHEAD + col] = (__bf16)(acc[dt][e] / l8[e]);
    }
}

// ---------------------------------------------------------------------------
// Kernel 4: out = attn(bf16) @ Wout + bout -> fp32. Grid (48, 16).
// ---------------------------------------------------------------------------
__global__ void __launch_bounds__(128)
out_gemm_kernel(const __bf16* __restrict__ A, const float* __restrict__ W,
                const float* __restrict__ bias, float* __restrict__ out) {
  __shared__ alignas(16) __bf16 As[64][32];
  __shared__ alignas(16) __bf16 Ws[32][64];
  const int t = threadIdx.x;
  const int wv = t >> 5, lane = t & 31, half = lane >> 4, l16 = lane & 15;
  const int m0 = blockIdx.x * 64;
  const int n0 = blockIdx.y * 64;

  f32x8 acc[4] = {};

  for (int k0 = 0; k0 < DMODEL; k0 += 32) {
    for (int i = 0; i < 2; i++) {              // A tile 64x32 bf16: pure copy
      int chunk = t + i * 128;
      int r = chunk >> 2, c8 = (chunk & 3) * 8;
      *(U4*)&As[r][c8] = *(const U4*)&A[(size_t)(m0 + r) * DMODEL + k0 + c8];
    }
    for (int i = 0; i < 2; i++) {              // W tile 32x64 f32 -> bf16
      int chunk = t + i * 128;
      int r = chunk >> 3, c8 = (chunk & 7) * 8;
      *(U4*)&Ws[r][c8] = cvt8(&W[(size_t)(k0 + r) * DMODEL + n0 + c8]);
    }
    __syncthreads();

    bf16x16 bfrag;
    for (int e = 0; e < 8; e++) bfrag[e]     = Ws[8 * half + e][wv * 16 + l16];
    for (int e = 0; e < 8; e++) bfrag[e + 8] = Ws[16 + 8 * half + e][wv * 16 + l16];
    for (int mt = 0; mt < 4; mt++) {
      bf16x16 afrag;
      int row = mt * 16 + l16;
      for (int e = 0; e < 8; e++) afrag[e]     = As[row][8 * half + e];
      for (int e = 0; e < 8; e++) afrag[e + 8] = As[row][16 + 8 * half + e];
      acc[mt] = wmma_bf16f32(afrag, bfrag, acc[mt]);
    }
    __syncthreads();
  }

  const int col = n0 + wv * 16 + l16;
  const float bc = bias[col];
  for (int mt = 0; mt < 4; mt++)
    for (int e = 0; e < 8; e++) {
      int row = m0 + mt * 16 + e + 8 * half;
      out[(size_t)row * DMODEL + col] = acc[mt][e] + bc;
    }
}

// ---------------------------------------------------------------------------
extern "C" void kernel_launch(void* const* d_in, const int* in_sizes, int n_in,
                              void* d_out, int out_size, void* d_ws, size_t ws_size,
                              hipStream_t stream) {
  const float* x    = (const float*)d_in[0];
  const float* Wqkv = (const float*)d_in[1];
  const float* W1   = (const float*)d_in[2];
  const float* b1   = (const float*)d_in[3];
  const float* W2   = (const float*)d_in[4];
  const float* b2   = (const float*)d_in[5];
  const float* Wout = (const float*)d_in[6];
  const float* bout = (const float*)d_in[7];
  float* out = (float*)d_out;

  const size_t NHD = (size_t)NHEADS * N_TOK * DHEAD;    // 3,145,728 elems
  char* ws = (char*)d_ws;
  __bf16* qkv = (__bf16*)ws;                            // 3*NHD bf16 = 18 MB
  __bf16* qb = qkv;
  __bf16* kb = qkv + NHD;
  __bf16* vb = qkv + 2 * NHD;
  float*  wscale = (float*)(ws + 3 * NHD * sizeof(__bf16));
  __bf16* attn   = (__bf16*)(ws + 3 * NHD * sizeof(__bf16)
                                + (size_t)NHEADS * N_TOK * sizeof(float));

  qkv_gemm_kernel  <<<dim3(48, 48), 128, 0, stream>>>(x, Wqkv, qkv);
  spiky_kernel     <<<16, 256, 0, stream>>>(kb, W1, b1, W2, b2, wscale);
  flash_attn_kernel<<<dim3(48, 16), 128, 0, stream>>>(qb, kb, vb, wscale, attn);
  out_gemm_kernel  <<<dim3(48, 16), 128, 0, stream>>>(attn, Wout, bout, out);
}
